// MultiHeadAttention_49246095016569
// MI455X (gfx1250) — compile-verified, matrix operand order checked by reading, not run on hardware
//
#include <hip/hip_runtime.h>
#include <hip/hip_bf16.h>

// MI455X (gfx1250) flash-attention MHA, bf16 WMMA everywhere with f32 accum.
// Layout insight: the reference's reshapes are flat reinterpretations, so each
// (b,h) is a contiguous row-major 2048x64 matrix, and the FC input is the flat
// [B,H,S,D] buffer viewed as 4096x512 row-major.

#define S_LEN 2048
#define BATCH 2
#define HEADS 8
#define HD 64
#define EMB 512
#define BH (BATCH * HEADS)   // 16
#define ROWS (S_LEN * BATCH) // 4096 FC rows

typedef __attribute__((ext_vector_type(16))) __bf16 v16bf;
typedef __attribute__((ext_vector_type(8)))  float  v8f;

// K-index within a 16x32 (A) / 32x16 (B) bf16 fragment for element i, lane-half h.
__device__ __forceinline__ int kidx(int i, int h) {
    return (i & 7) | ((i & 8) << 1) | (h << 3);
}

__device__ __forceinline__ float hmax16(float v) {
#pragma unroll
    for (int m = 1; m < 16; m <<= 1) v = fmaxf(v, __shfl_xor(v, m, 32));
    return v;
}
__device__ __forceinline__ float hsum16(float v) {
#pragma unroll
    for (int m = 1; m < 16; m <<= 1) v += __shfl_xor(v, m, 32);
    return v;
}

__device__ __forceinline__ v8f wmma_bf16(v16bf a, v16bf b, v8f c) {
    return __builtin_amdgcn_wmma_f32_16x16x32_bf16(
        /*neg_a=*/false, a, /*neg_b=*/false, b,
        /*c_mod=*/(short)0, c, /*reuse_a=*/false, /*reuse_b=*/false);
}

// ---------------------------------------------------------------- Wfc -> bf16
__global__ __launch_bounds__(256) void cvt_bf16_kernel(const float* __restrict__ w,
                                                       __bf16* __restrict__ wb, int n) {
    int i = blockIdx.x * 256 + threadIdx.x;
    if (i < n) wb[i] = (__bf16)w[i];
}

// ------------------------------------------------- Q/K/V projection (X @ W^T)
// grid.x = BH * (2048/128), grid.y selects tensor (0=Q scaled, 1=K, 2=V).
// 8 waves/block, each wave produces a 16x64 bf16 output tile via 8 WMMAs.
__global__ __launch_bounds__(256) void proj_kernel(
    const float* __restrict__ q, const float* __restrict__ k, const float* __restrict__ v,
    const float* __restrict__ Wq, const float* __restrict__ Wk, const float* __restrict__ Wv,
    __bf16* __restrict__ Qp, __bf16* __restrict__ Kp, __bf16* __restrict__ Vp) {
    const float* X; const float* W; __bf16* O; float scale;
    if (blockIdx.y == 0)      { X = q; W = Wq; O = Qp; scale = 0.125f; }  // 1/sqrt(64)
    else if (blockIdx.y == 1) { X = k; W = Wk; O = Kp; scale = 1.0f; }
    else                      { X = v; W = Wv; O = Vp; scale = 1.0f; }

    const int lane = threadIdx.x & 31, wid = threadIdx.x >> 5;
    const int half = lane >> 4, n = lane & 15;
    const int bh = blockIdx.x >> 4, rowblk = blockIdx.x & 15;
    const int m0 = rowblk * 128 + wid * 16;

    const float* Xrow = X + (size_t)(bh * S_LEN + m0 + n) * HD;
    v16bf a0, a1;
#pragma unroll
    for (int i = 0; i < 16; ++i) {
        int e = kidx(i, half);
        a0[i] = (__bf16)Xrow[e];
        a1[i] = (__bf16)Xrow[32 + e];
    }
#pragma unroll
    for (int nb = 0; nb < 4; ++nb) {
        const float* Wrow = W + (nb * 16 + n) * HD;  // B[k=e][n=d] = W[d][e]
        v16bf b0, b1;
#pragma unroll
        for (int i = 0; i < 16; ++i) {
            int e = kidx(i, half);
            b0[i] = (__bf16)Wrow[e];
            b1[i] = (__bf16)Wrow[32 + e];
        }
        v8f c = {};
        c = wmma_bf16(a0, b0, c);
        c = wmma_bf16(a1, b1, c);
#pragma unroll
        for (int r = 0; r < 8; ++r) {
            int m = m0 + r + 8 * half;
            O[(size_t)(bh * S_LEN + m) * HD + nb * 16 + n] = (__bf16)(c[r] * scale);
        }
    }
}

// -------------------------------------------------------- flash attention core
// grid.x = BH * (2048/128). 8 waves/block; wave owns 16 query rows x full D=64.
// Key tiles of 32 staged in LDS; online softmax in f32; P transposed via LDS.
__global__ __launch_bounds__(256) void attn_kernel(
    const __bf16* __restrict__ Qp, const __bf16* __restrict__ Kp,
    const __bf16* __restrict__ Vp, __bf16* __restrict__ Ap) {
    __shared__ __bf16 Ks[32 * 64];
    __shared__ __bf16 Vs[32 * 64];
    __shared__ __bf16 Psh[8 * 16 * 32];

    const int tid = threadIdx.x;
    const int lane = tid & 31, wid = tid >> 5;
    const int half = lane >> 4, n = lane & 15;
    const int bh = blockIdx.x >> 4, qblk = blockIdx.x & 15;
    const int m0 = qblk * 128 + wid * 16;

    const __bf16* Qrow = Qp + (size_t)(bh * S_LEN + m0 + n) * HD;
    v16bf qa0, qa1;
#pragma unroll
    for (int i = 0; i < 16; ++i) {
        int e = kidx(i, half);
        qa0[i] = Qrow[e];
        qa1[i] = Qrow[32 + e];
    }

    v8f o0 = {}, o1 = {}, o2 = {}, o3 = {};
    float mrow[8], lrow[8];
#pragma unroll
    for (int r = 0; r < 8; ++r) { mrow[r] = -1e30f; lrow[r] = 0.0f; }

    const int crow = tid >> 3;          // cooperative-copy: 32 rows of K/V tile
    const int cd = (tid & 7) * 8;       // 8 bf16 = 16B per thread per tensor

    for (int t = 0; t < 64; ++t) {
        const size_t gbase = (size_t)(bh * S_LEN + t * 32 + crow) * HD + cd;
        *(uint4*)&Ks[crow * 64 + cd] = *(const uint4*)&Kp[gbase];
        *(uint4*)&Vs[crow * 64 + cd] = *(const uint4*)&Vp[gbase];
        if (t + 1 < 64)  // global_prefetch_b8 of next tile
            __builtin_prefetch(&Kp[gbase + 32 * HD], 0, 0);
        __syncthreads();

        // S = Q @ K^T for this 16x32 score tile (two 16x16 column halves)
        v8f s0 = {}, s1 = {};
        {
            v16bf kb0, kb1;
#pragma unroll
            for (int i = 0; i < 16; ++i) {
                int e = kidx(i, half);
                kb0[i] = Ks[n * 64 + e];
                kb1[i] = Ks[n * 64 + 32 + e];
            }
            s0 = wmma_bf16(qa0, kb0, s0);
            s0 = wmma_bf16(qa1, kb1, s0);
#pragma unroll
            for (int i = 0; i < 16; ++i) {
                int e = kidx(i, half);
                kb0[i] = Ks[(16 + n) * 64 + e];
                kb1[i] = Ks[(16 + n) * 64 + 32 + e];
            }
            s1 = wmma_bf16(qa0, kb0, s1);
            s1 = wmma_bf16(qa1, kb1, s1);
        }

        // online softmax: row M = r + 8*half lives on one 16-lane half
#pragma unroll
        for (int r = 0; r < 8; ++r) {
            float mt = hmax16(fmaxf(s0[r], s1[r]));
            float mnew = fmaxf(mrow[r], mt);
            float alpha = __expf(mrow[r] - mnew);
            float p0 = __expf(s0[r] - mnew);
            float p1 = __expf(s1[r] - mnew);
            lrow[r] = lrow[r] * alpha + hsum16(p0 + p1);
            mrow[r] = mnew;
            o0[r] *= alpha; o1[r] *= alpha; o2[r] *= alpha; o3[r] *= alpha;
            int m = r + 8 * half;
            Psh[wid * 512 + m * 32 + n]      = (__bf16)p0;
            Psh[wid * 512 + m * 32 + 16 + n] = (__bf16)p1;
        }

        // re-fragment P (C-layout -> A-layout) through wave-private LDS
        v16bf pa;
#pragma unroll
        for (int i = 0; i < 16; ++i)
            pa[i] = Psh[wid * 512 + n * 32 + kidx(i, half)];

        // O += P @ V
#pragma unroll
        for (int nb = 0; nb < 4; ++nb) {
            v16bf vb;
#pragma unroll
            for (int i = 0; i < 16; ++i)
                vb[i] = Vs[kidx(i, half) * 64 + nb * 16 + n];
            if (nb == 0) o0 = wmma_bf16(pa, vb, o0);
            else if (nb == 1) o1 = wmma_bf16(pa, vb, o1);
            else if (nb == 2) o2 = wmma_bf16(pa, vb, o2);
            else o3 = wmma_bf16(pa, vb, o3);
        }
        __syncthreads();
    }

#pragma unroll
    for (int r = 0; r < 8; ++r) {
        float inv = 1.0f / lrow[r];
        int m = m0 + r + 8 * half;
        size_t base = (size_t)(bh * S_LEN + m) * HD;
        Ap[base + 0 * 16 + n] = (__bf16)(o0[r] * inv);
        Ap[base + 1 * 16 + n] = (__bf16)(o1[r] * inv);
        Ap[base + 2 * 16 + n] = (__bf16)(o2[r] * inv);
        Ap[base + 3 * 16 + n] = (__bf16)(o3[r] * inv);
    }
}

// ------------------------------------------------------ final FC: att@Wfc^T+b
// A = Ap viewed as 4096x512 row-major (flat [B,H,S,D] == FC input rows).
// grid = (4096/32, 512/64); 8 waves/block -> 2 row-tiles x 4 col-tiles.
__global__ __launch_bounds__(256) void fc_kernel(
    const __bf16* __restrict__ Ap, const __bf16* __restrict__ Wf,
    const float* __restrict__ bfc, float* __restrict__ out) {
    const int lane = threadIdx.x & 31, wid = threadIdx.x >> 5;
    const int half = lane >> 4, n = lane & 15;
    const int m0 = blockIdx.x * 32 + (wid >> 2) * 16;
    const int n0 = blockIdx.y * 64 + (wid & 3) * 16;

    v8f c = {};
    const __bf16* Arow = Ap + (size_t)(m0 + n) * EMB;
    const __bf16* Brow = Wf + (size_t)(n0 + n) * EMB;  // B[k=e][n=j] = Wfc[j][e]
#pragma unroll 4
    for (int kt = 0; kt < 16; ++kt) {
        int kb = kt * 32;
        v16bf a, b;
#pragma unroll
        for (int i = 0; i < 16; ++i) {
            int e = kidx(i, half);
            a[i] = Arow[kb + e];
            b[i] = Brow[kb + e];
        }
        c = wmma_bf16(a, b, c);
    }
    float bias = bfc[n0 + n];
#pragma unroll
    for (int r = 0; r < 8; ++r)
        out[(size_t)(m0 + r + 8 * half) * EMB + n0 + n] = c[r] + bias;
}

extern "C" void kernel_launch(void* const* d_in, const int* in_sizes, int n_in,
                              void* d_out, int out_size, void* d_ws, size_t ws_size,
                              hipStream_t stream) {
    const float* q   = (const float*)d_in[0];
    const float* k   = (const float*)d_in[1];
    const float* v   = (const float*)d_in[2];
    const float* Wq  = (const float*)d_in[3];
    const float* Wk  = (const float*)d_in[4];
    const float* Wv  = (const float*)d_in[5];
    const float* Wfc = (const float*)d_in[6];
    const float* bfc = (const float*)d_in[7];
    float* out = (float*)d_out;

    const size_t HBUF = (size_t)BH * S_LEN * HD;  // 2,097,152 bf16 elems = 4 MiB
    char* ws = (char*)d_ws;
    __bf16* Qp  = (__bf16*)(ws + 0 * HBUF * 2);
    __bf16* Kp  = (__bf16*)(ws + 1 * HBUF * 2);
    __bf16* Vp  = (__bf16*)(ws + 2 * HBUF * 2);
    __bf16* Ap  = (__bf16*)(ws + 3 * HBUF * 2);
    __bf16* Wfb = (__bf16*)(ws + 4 * HBUF * 2);   // 512 KiB; total ~16.5 MiB

    cvt_bf16_kernel<<<(EMB * EMB + 255) / 256, 256, 0, stream>>>(Wfc, Wfb, EMB * EMB);
    proj_kernel<<<dim3(BH * (S_LEN / 128), 3), 256, 0, stream>>>(
        q, k, v, Wq, Wk, Wv, Qp, Kp, Vp);
    attn_kernel<<<BH * (S_LEN / 128), 256, 0, stream>>>(Qp, Kp, Vp, Ap);
    fc_kernel<<<dim3(ROWS / 32, EMB / 64), 256, 0, stream>>>(Ap, Wfb, bfc, out);
}
